// backbone_pointnet2_77704548319755
// MI455X (gfx1250) — matrix operand order repaired and checked by reading, not compiled, not run on hardware
//
#include <hip/hip_runtime.h>
#include <hip/hip_bf16.h>
#include <cstdint>

typedef __bf16 bf16;
typedef __attribute__((ext_vector_type(8)))  __bf16 v8bf;
typedef __attribute__((ext_vector_type(16))) __bf16 v16bf;
typedef __attribute__((ext_vector_type(8)))  float  v8f;

#define BB 4
#define NN 16384
#define TT 5
#define EPS_BN 1e-5f

static inline int cdiv_h(long a, long b) { return (int)((a + b - 1) / b); }
static inline int ceil64_h(int x) { return (x + 63) & ~63; }

// ---------------------------------------------------------------------------
// Prep: transpose points -> feats0 (B,N,3) bf16 and uf1 = [xyz|feats0] (B,N,6)
// ---------------------------------------------------------------------------
__global__ void prep_feats_kernel(const float* __restrict__ points,
                                  const float* __restrict__ xyz,
                                  bf16* __restrict__ feats0,
                                  bf16* __restrict__ uf1)
{
  long i = (long)blockIdx.x * blockDim.x + threadIdx.x; // over B*N
  if (i >= (long)BB * NN) return;
  long b = i / NN, n = i % NN;
  float f0 = points[(b * 3 + 0) * NN + n];
  float f1 = points[(b * 3 + 1) * NN + n];
  float f2 = points[(b * 3 + 2) * NN + n];
  feats0[i * 3 + 0] = (bf16)f0;
  feats0[i * 3 + 1] = (bf16)f1;
  feats0[i * 3 + 2] = (bf16)f2;
  uf1[i * 6 + 0] = (bf16)xyz[i * 3 + 0];
  uf1[i * 6 + 1] = (bf16)xyz[i * 3 + 1];
  uf1[i * 6 + 2] = (bf16)xyz[i * 3 + 2];
  uf1[i * 6 + 3] = (bf16)f0;
  uf1[i * 6 + 4] = (bf16)f1;
  uf1[i * 6 + 5] = (bf16)f2;
}

// ---------------------------------------------------------------------------
// Farthest point sampling: one 1024-thread block per batch, dists in registers
// ---------------------------------------------------------------------------
__global__ __launch_bounds__(1024)
void fps_kernel(const float* __restrict__ xyz, int n, int npoint,
                int* __restrict__ fidx)
{
  int b = blockIdx.x;
  const float* P = xyz + (size_t)b * n * 3;
  int* out = fidx + (size_t)b * npoint;
  int tid = threadIdx.x;
  const int PT = 16;
  float dist[PT];
  int cnt = (n + 1023) >> 10;
#pragma unroll
  for (int i = 0; i < PT; i++) dist[i] = 1e10f;

  __shared__ float sval[32];
  __shared__ int   sidx[32];
  __shared__ int   s_last;
  if (tid == 0) { s_last = 0; out[0] = 0; }
  __syncthreads();

  for (int it = 1; it < npoint; ++it) {
    int last = s_last;
    float lx = P[last * 3 + 0], ly = P[last * 3 + 1], lz = P[last * 3 + 2];
    float best = -1.0f; int bi = 0;
    for (int i = 0; i < cnt; i++) {
      int p = tid + (i << 10);
      if (p < n) {
        float dx = P[p * 3 + 0] - lx;
        float dy = P[p * 3 + 1] - ly;
        float dz = P[p * 3 + 2] - lz;
        float d = dx * dx + dy * dy + dz * dz;
        float dm = fminf(dist[i], d);
        dist[i] = dm;
        if (dm > best) { best = dm; bi = p; }
      }
    }
    // wave32 shuffle reduce
    for (int off = 16; off > 0; off >>= 1) {
      float ov = __shfl_down(best, off, 32);
      int   oi = __shfl_down(bi, off, 32);
      if (ov > best) { best = ov; bi = oi; }
    }
    int wave = tid >> 5;
    if ((tid & 31) == 0) { sval[wave] = best; sidx[wave] = bi; }
    __syncthreads();
    if (tid < 32) {
      best = sval[tid]; bi = sidx[tid];
      for (int off = 16; off > 0; off >>= 1) {
        float ov = __shfl_down(best, off, 32);
        int   oi = __shfl_down(bi, off, 32);
        if (ov > best) { best = ov; bi = oi; }
      }
      if (tid == 0) { s_last = bi; out[it] = bi; }
    }
    __syncthreads();
  }
}

__global__ void gather_xyz_kernel(const float* __restrict__ xyz,
                                  const int* __restrict__ fidx,
                                  int n, int m, float* __restrict__ nx)
{
  int g = blockIdx.x * blockDim.x + threadIdx.x;
  if (g >= BB * m) return;
  int b = g / m, j = g % m;
  int p = fidx[(size_t)b * m + j];
  const float* s = xyz + ((size_t)b * n + p) * 3;
  nx[(size_t)g * 3 + 0] = s[0];
  nx[(size_t)g * 3 + 1] = s[1];
  nx[(size_t)g * 3 + 2] = s[2];
}

// ---------------------------------------------------------------------------
// Ball query: first nsample indices (ascending) within radius, fallback fill
// ---------------------------------------------------------------------------
__global__ void ball_query_kernel(const float* __restrict__ xyz,
                                  const float* __restrict__ nxyz,
                                  int n, int m, float r2, int nsample,
                                  int* __restrict__ idx)
{
  int g = blockIdx.x * blockDim.x + threadIdx.x;
  if (g >= BB * m) return;
  int b = g / m;
  const float* P = xyz + (size_t)b * n * 3;
  float qx = nxyz[(size_t)g * 3 + 0];
  float qy = nxyz[(size_t)g * 3 + 1];
  float qz = nxyz[(size_t)g * 3 + 2];
  int* o = idx + (size_t)g * nsample;
  int cnt = 0;
  for (int p = 0; p < n && cnt < nsample; p++) {
    float dx = P[p * 3 + 0] - qx;
    float dy = P[p * 3 + 1] - qy;
    float dz = P[p * 3 + 2] - qz;
    if (dx * dx + dy * dy + dz * dz < r2) o[cnt++] = p;
  }
  int f = (cnt > 0) ? o[0] : 0;
  for (; cnt < nsample; cnt++) o[cnt] = f;
}

// ---------------------------------------------------------------------------
// Build grouped GEMM input rows: [xyz[idx]-center | feats[idx] | 0-pad] bf16
// ---------------------------------------------------------------------------
__global__ void group_gather_kernel(const float* __restrict__ xyz,
                                    const float* __restrict__ nxyz,
                                    const bf16* __restrict__ feats,
                                    const int* __restrict__ idx,
                                    int n, int m, int S, int Cf, int Cld, int Kp,
                                    bf16* __restrict__ X)
{
  long r = (long)blockIdx.x * blockDim.x + threadIdx.x;
  long total = (long)BB * m * S;
  if (r >= total) return;
  int b  = (int)(r / ((long)m * S));
  int js = (int)(r % ((long)m * S));
  int j  = js / S;
  int pid = idx[((size_t)b * m + j) * S + (js % S)];
  const float* pp = xyz + ((size_t)b * n + pid) * 3;
  const float* cc = nxyz + ((size_t)b * m + j) * 3;
  bf16* xr = X + (size_t)r * Kp;
  xr[0] = (bf16)(pp[0] - cc[0]);
  xr[1] = (bf16)(pp[1] - cc[1]);
  xr[2] = (bf16)(pp[2] - cc[2]);
  const bf16* fr = feats + ((size_t)b * n + pid) * Cld;
  for (int c = 0; c < Cf; c++) xr[3 + c] = fr[c];
  for (int c = 3 + Cf; c < Kp; c++) xr[c] = (bf16)0.f;
}

__global__ void sa_all_concat_kernel(const float* __restrict__ x3,
                                     const bf16* __restrict__ f3,
                                     int rows, int Cf, int Cld, int Kp,
                                     bf16* __restrict__ X)
{
  long r = (long)blockIdx.x * blockDim.x + threadIdx.x;
  if (r >= rows) return;
  bf16* xr = X + (size_t)r * Kp;
  xr[0] = (bf16)x3[r * 3 + 0];
  xr[1] = (bf16)x3[r * 3 + 1];
  xr[2] = (bf16)x3[r * 3 + 2];
  const bf16* fr = f3 + (size_t)r * Cld;
  for (int c = 0; c < Cf; c++) xr[3 + c] = fr[c];
  for (int c = 3 + Cf; c < Kp; c++) xr[c] = (bf16)0.f;
}

__global__ void fp4_concat_kernel(const bf16* __restrict__ l4f,
                                  const bf16* __restrict__ l3f,
                                  bf16* __restrict__ X)
{
  long i = (long)blockIdx.x * blockDim.x + threadIdx.x; // rows 256, K 768
  if (i >= 256L * 768) return;
  int r = (int)(i / 768), c = (int)(i % 768);
  int b = r / 64;
  X[i] = (c < 512) ? l4f[(size_t)b * 512 + c]
                   : l3f[(size_t)r * 256 + (c - 512)];
}

// ---------------------------------------------------------------------------
// FP 3-NN inverse-distance interpolation + concat with skip features
// ---------------------------------------------------------------------------
__global__ void fp_interp_kernel(const float* __restrict__ ux,
                                 const float* __restrict__ kx,
                                 const bf16* __restrict__ kf,
                                 const bf16* __restrict__ uf,
                                 int nu, int nk, int Cf, int Cfld,
                                 int Cu, int Culd, int Kp,
                                 bf16* __restrict__ X)
{
  int g = blockIdx.x * blockDim.x + threadIdx.x;
  if (g >= BB * nu) return;
  int b = g / nu;
  float px = ux[(size_t)g * 3 + 0];
  float py = ux[(size_t)g * 3 + 1];
  float pz = ux[(size_t)g * 3 + 2];
  const float* K = kx + (size_t)b * nk * 3;
  float d0 = 3e38f, d1 = 3e38f, d2 = 3e38f;
  int i0 = 0, i1 = 0, i2 = 0;
  for (int p = 0; p < nk; p++) {
    float dx = K[p * 3 + 0] - px;
    float dy = K[p * 3 + 1] - py;
    float dz = K[p * 3 + 2] - pz;
    float d = dx * dx + dy * dy + dz * dz;
    if (d < d0)      { d2 = d1; i2 = i1; d1 = d0; i1 = i0; d0 = d; i0 = p; }
    else if (d < d1) { d2 = d1; i2 = i1; d1 = d; i1 = p; }
    else if (d < d2) { d2 = d; i2 = p; }
  }
  float w0 = 1.f / (d0 + 1e-8f), w1 = 1.f / (d1 + 1e-8f), w2 = 1.f / (d2 + 1e-8f);
  float ws = w0 + w1 + w2;
  w0 /= ws; w1 /= ws; w2 /= ws;
  bf16* xr = X + (size_t)g * Kp;
  const bf16* f0 = kf + ((size_t)b * nk + i0) * Cfld;
  const bf16* f1 = kf + ((size_t)b * nk + i1) * Cfld;
  const bf16* f2 = kf + ((size_t)b * nk + i2) * Cfld;
  for (int c = 0; c < Cf; c++)
    xr[c] = (bf16)(w0 * (float)f0[c] + w1 * (float)f1[c] + w2 * (float)f2[c]);
  const bf16* u = uf + (size_t)g * Culd;
  for (int c = 0; c < Cu; c++) xr[Cf + c] = u[c];
  for (int c = Cf + Cu; c < Kp; c++) xr[c] = (bf16)0.f;
}

// ---------------------------------------------------------------------------
// Weight prep: f32 (O,K) -> bf16 (Op,Kp) zero-padded
// ---------------------------------------------------------------------------
__global__ void wprep_kernel(const float* __restrict__ W, int O, int K,
                             int Op, int Kp, bf16* __restrict__ Wb)
{
  long i = (long)blockIdx.x * blockDim.x + threadIdx.x;
  if (i >= (long)Op * Kp) return;
  int o = (int)(i / Kp), k = (int)(i % Kp);
  Wb[i] = (o < O && k < K) ? (bf16)W[(size_t)o * K + k] : (bf16)0.f;
}

__global__ void fill0_kernel(float* __restrict__ p, int n)
{
  int i = blockIdx.x * blockDim.x + threadIdx.x;
  if (i < n) p[i] = 0.f;
}

// ---------------------------------------------------------------------------
// Core: bf16 WMMA GEMM  Y[m,o] = sum_k X[m,k]*W[o,k]
//   block = 4 waves, tile M=64 x O=64, K step 32 (v_wmma_f32_16x16x32_bf16)
//   W tile staged into LDS with global_load_async_to_lds_b128 (TENSOR/ASYNC path)
//   A fragments read straight from global (L2-resident activations)
//   Optional per-channel sum/sumsq (batchnorm stats) via LDS+global atomics
// ---------------------------------------------------------------------------
#define WT_LD 40  // bf16 elements per LDS W-tile row (32 data + 8 pad)

__global__ __launch_bounds__(128)
void gemm_bf16_kernel(const bf16* __restrict__ X, const bf16* __restrict__ Wt,
                      bf16* __restrict__ Ybf, float* __restrict__ Yf,
                      int M, int Kp, int Op, float* __restrict__ stats)
{
  __shared__ __align__(16) bf16 wt[64 * WT_LD];
  __shared__ float ssum[64];
  __shared__ float ssq[64];

  const int tid   = threadIdx.x;
  const int lane  = tid & 31;
  const int wave  = tid >> 5;
  const int lrow  = lane & 15;
  const int lhalf = lane >> 4;
  const int m0 = blockIdx.x * 64 + wave * 16;
  const int o0 = blockIdx.y * 64;

  v8f zero = {0.f, 0.f, 0.f, 0.f, 0.f, 0.f, 0.f, 0.f};
  v8f acc[4];
#pragma unroll
  for (int j = 0; j < 4; ++j) acc[j] = zero;

  const bf16* xbase = X + (size_t)(m0 + lrow) * Kp;
  __builtin_prefetch(xbase, 0, 0);

  for (int k0 = 0; k0 < Kp; k0 += 32) {
    __syncthreads(); // previous tile fully consumed
    // ---- async stage of W tile (64 rows x 32 bf16) into LDS ----
#pragma unroll
    for (int c = tid; c < 256; c += 128) {
      int row = c >> 2, piece = c & 3;
      uint32_t lds = (uint32_t)(uintptr_t)(void*)&wt[row * WT_LD + piece * 8];
      const bf16* gp = Wt + (size_t)(o0 + row) * Kp + k0 + piece * 8;
      asm volatile("global_load_async_to_lds_b128 %0, %1, off"
                   :: "v"(lds), "v"(gp) : "memory");
    }
    asm volatile("s_wait_asynccnt 0x0" ::: "memory");
    __syncthreads();

    // ---- A fragment: per ISA 16-bit A 16x32 layout ----
    const bf16* xr = xbase + k0 + lhalf * 8;
    v8bf a0 = *(const v8bf*)(xr);
    v8bf a1 = *(const v8bf*)(xr + 16);
    v16bf a = __builtin_shufflevector(a0, a1, 0, 1, 2, 3, 4, 5, 6, 7,
                                      8, 9, 10, 11, 12, 13, 14, 15);
    // ---- 4 column sub-tiles ----
#pragma unroll
    for (int j = 0; j < 4; ++j) {
      const bf16* wr = &wt[(j * 16 + lrow) * WT_LD + lhalf * 8];
      v8bf b0 = *(const v8bf*)(wr);
      v8bf b1 = *(const v8bf*)(wr + 16);
      v16bf bb = __builtin_shufflevector(b0, b1, 0, 1, 2, 3, 4, 5, 6, 7,
                                         8, 9, 10, 11, 12, 13, 14, 15);
      acc[j] = __builtin_amdgcn_wmma_f32_16x16x32_bf16(
          false, a, false, bb, (short)0, acc[j], false, false);
    }
  }

  if (stats) {
    if (tid < 64) { ssum[tid] = 0.f; ssq[tid] = 0.f; }
    __syncthreads();
  }

#pragma unroll
  for (int j = 0; j < 4; ++j) {
    int o = o0 + j * 16 + lrow;
    float s = 0.f, q = 0.f;
#pragma unroll
    for (int r = 0; r < 8; ++r) {
      float v = acc[j][r];
      int m = m0 + lhalf * 8 + r;
      size_t off = (size_t)m * Op + o;
      if (Ybf) Ybf[off] = (bf16)v;
      if (Yf)  Yf[off] = v;
      s += v; q += v * v;
    }
    if (stats) {
      atomicAdd(&ssum[j * 16 + lrow], s);  // ds_add_f32
      atomicAdd(&ssq[j * 16 + lrow], q);
    }
  }
  if (stats) {
    __syncthreads();
    if (tid < 64) {
      atomicAdd(&stats[o0 + tid], ssum[tid]);
      atomicAdd(&stats[Op + o0 + tid], ssq[tid]);
    }
  }
}

// ---------------------------------------------------------------------------
// BatchNorm finalize + apply (training-mode stats over all rows)
// ---------------------------------------------------------------------------
__global__ void bn_finalize_kernel(const float* __restrict__ stats,
                                   const float* __restrict__ g,
                                   const float* __restrict__ bt,
                                   int O, int Op, float cnt,
                                   float* __restrict__ sc, float* __restrict__ sh)
{
  int o = blockIdx.x * blockDim.x + threadIdx.x;
  if (o >= Op) return;
  if (o < O) {
    float mean = stats[o] / cnt;
    float var = stats[Op + o] / cnt - mean * mean;
    float s = g[o] * rsqrtf(var + EPS_BN);
    sc[o] = s;
    sh[o] = bt[o] - mean * s;
  } else { sc[o] = 0.f; sh[o] = 0.f; }
}

__global__ void bn_act_kernel(const bf16* __restrict__ Y,
                              const float* __restrict__ sc,
                              const float* __restrict__ sh,
                              float slope, long total, int Op,
                              bf16* __restrict__ Xo)
{
  long i = (long)blockIdx.x * blockDim.x + threadIdx.x;
  if (i >= total) return;
  int o = (int)(i % Op);
  float v = (float)Y[i] * sc[o] + sh[o];
  v = (v > 0.f) ? v : slope * v;
  Xo[i] = (bf16)v;
}

__global__ void maxpool_kernel(const bf16* __restrict__ X, int S, int Op,
                               long G, bf16* __restrict__ out)
{
  long i = (long)blockIdx.x * blockDim.x + threadIdx.x;
  if (i >= G * Op) return;
  long g = i / Op; int o = (int)(i % Op);
  const bf16* p = X + (size_t)g * S * Op + o;
  float m = -3e38f;
  for (int s = 0; s < S; s++) m = fmaxf(m, (float)p[(size_t)s * Op]);
  out[i] = (bf16)m;
}

// ---------------------------------------------------------------------------
// Final: scatter conv3 f32 output (B*N,135 @ ld 192) into the 3-way split
// ---------------------------------------------------------------------------
__global__ void out_scatter_kernel(const float* __restrict__ Yf,
                                   float* __restrict__ out)
{
  long i = (long)blockIdx.x * blockDim.x + threadIdx.x;
  long total = (long)BB * NN * 135;
  if (i >= total) return;
  long r = i / 135; int o = (int)(i % 135);
  float v = Yf[r * 192 + o];
  int c = o / TT, t = o % TT;
  long base; long C;
  if (c < 2)      { base = 0;                         C = 2;  }
  else if (c < 7) { base = (long)BB * NN * 2 * TT;    C = 5;  c -= 2; }
  else            { base = (long)BB * NN * 7 * TT;    C = 20; c -= 7; }
  out[base + (r * C + c) * TT + t] = v;
}

// ===========================================================================
// Host orchestration
// ===========================================================================
extern "C" void kernel_launch(void* const* d_in, const int* in_sizes, int n_in,
                              void* d_out, int out_size, void* d_ws, size_t ws_size,
                              hipStream_t stream)
{
  (void)in_sizes; (void)n_in; (void)out_size; (void)ws_size;
  const float* xyz    = (const float*)d_in[0];
  const float* points = (const float*)d_in[1];

  struct Lyr { const float *W, *g, *b; int K, O; };
  static const int Kc[22] = { 6, 32, 32,  67, 64, 64,  131, 128, 128,
                              259, 256, 256,  768, 256,  384, 256,
                              320, 256,  134, 128,  128, 128 };
  static const int Oc[22] = { 32, 32, 64,  64, 64, 128,  128, 128, 256,
                              256, 256, 512,  256, 256,  256, 256,
                              256, 128,  128, 128,  128, 128 };
  Lyr L[22];
  int pi = 2;
  for (int i = 0; i < 22; i++) {
    L[i].W = (const float*)d_in[pi++];
    L[i].g = (const float*)d_in[pi++];
    L[i].b = (const float*)d_in[pi++];
    L[i].K = Kc[i]; L[i].O = Oc[i];
  }
  const float* conv3W = (const float*)d_in[pi++];

  // ---- workspace bump allocator ----
  char* wsp = (char*)d_ws;
  size_t off = 0;
  auto take = [&](size_t bytes) -> void* {
    void* p = wsp + off;
    off += (bytes + 255) & ~(size_t)255;
    return p;
  };
  bf16*  feats0 = (bf16*)take((size_t)BB * NN * 3 * 2);
  bf16*  uf1    = (bf16*)take((size_t)BB * NN * 6 * 2);
  float* l1x = (float*)take((size_t)BB * 1024 * 3 * 4);
  float* l2x = (float*)take((size_t)BB * 256 * 3 * 4);
  float* l3x = (float*)take((size_t)BB * 64 * 3 * 4);
  int* fidx1 = (int*)take((size_t)BB * 1024 * 4);
  int* fidx2 = (int*)take((size_t)BB * 256 * 4);
  int* fidx3 = (int*)take((size_t)BB * 64 * 4);
  int* idx1 = (int*)take((size_t)BB * 1024 * 32 * 4);
  int* idx2 = (int*)take((size_t)BB * 256 * 64 * 4);
  int* idx3 = (int*)take((size_t)BB * 64 * 128 * 4);
  bf16* l1f  = (bf16*)take((size_t)BB * 1024 * 64 * 2);
  bf16* l2f  = (bf16*)take((size_t)BB * 256 * 128 * 2);
  bf16* l3f  = (bf16*)take((size_t)BB * 64 * 256 * 2);
  bf16* l4f  = (bf16*)take((size_t)BB * 512 * 2);
  bf16* l3f2 = (bf16*)take((size_t)BB * 64 * 256 * 2);
  bf16* l2f2 = (bf16*)take((size_t)BB * 256 * 256 * 2);
  bf16* l1f2 = (bf16*)take((size_t)BB * 1024 * 128 * 2);
  bf16* XA = (bf16*)take((size_t)65536 * 192 * 2);
  bf16* YB = (bf16*)take((size_t)65536 * 192 * 2);
  float* Yf32 = (float*)take((size_t)65536 * 192 * 4);
  bf16* Wb = (bf16*)take((size_t)768 * 256 * 2);
  float* stats = (float*)take(2 * 512 * 4);
  float* sc = (float*)take(512 * 4);
  float* sh = (float*)take(512 * 4);

  auto run_layer = [&](const Lyr& ly, const bf16* Xin, bf16* Ytmp, bf16* Xout,
                       int M, float slope) {
    int Kp = ceil64_h(ly.K), Op = ceil64_h(ly.O);
    long wn = (long)Op * Kp;
    wprep_kernel<<<cdiv_h(wn, 256), 256, 0, stream>>>(ly.W, ly.O, ly.K, Op, Kp, Wb);
    fill0_kernel<<<cdiv_h(2 * Op, 256), 256, 0, stream>>>(stats, 2 * Op);
    dim3 g(M / 64, Op / 64);
    gemm_bf16_kernel<<<g, 128, 0, stream>>>(Xin, Wb, Ytmp, nullptr, M, Kp, Op, stats);
    bn_finalize_kernel<<<cdiv_h(Op, 64), 64, 0, stream>>>(stats, ly.g, ly.b,
                                                          ly.O, Op, (float)M, sc, sh);
    long tot = (long)M * Op;
    bn_act_kernel<<<cdiv_h(tot, 256), 256, 0, stream>>>(Ytmp, sc, sh, slope, tot, Op, Xout);
  };

  // ---- prep ----
  prep_feats_kernel<<<cdiv_h((long)BB * NN, 256), 256, 0, stream>>>(points, xyz, feats0, uf1);

  // ---- SA1: 16384 -> 1024, r=0.1, S=32, MLP[32,32,64] ----
  fps_kernel<<<BB, 1024, 0, stream>>>(xyz, NN, 1024, fidx1);
  gather_xyz_kernel<<<cdiv_h(BB * 1024, 256), 256, 0, stream>>>(xyz, fidx1, NN, 1024, l1x);
  ball_query_kernel<<<cdiv_h(BB * 1024, 128), 128, 0, stream>>>(xyz, l1x, NN, 1024,
                                                                0.01f, 32, idx1);
  group_gather_kernel<<<cdiv_h((long)BB * 1024 * 32, 256), 256, 0, stream>>>(
      xyz, l1x, feats0, idx1, NN, 1024, 32, 3, 3, 64, XA);
  run_layer(L[0], XA, YB, XA, 131072, 0.f);
  run_layer(L[1], XA, YB, XA, 131072, 0.f);
  run_layer(L[2], XA, YB, XA, 131072, 0.f);
  maxpool_kernel<<<cdiv_h((long)4096 * 64, 256), 256, 0, stream>>>(XA, 32, 64, 4096, l1f);

  // ---- SA2: 1024 -> 256, r=0.2, S=64, MLP[64,64,128] ----
  fps_kernel<<<BB, 1024, 0, stream>>>(l1x, 1024, 256, fidx2);
  gather_xyz_kernel<<<cdiv_h(BB * 256, 256), 256, 0, stream>>>(l1x, fidx2, 1024, 256, l2x);
  ball_query_kernel<<<cdiv_h(BB * 256, 128), 128, 0, stream>>>(l1x, l2x, 1024, 256,
                                                               0.04f, 64, idx2);
  group_gather_kernel<<<cdiv_h((long)BB * 256 * 64, 256), 256, 0, stream>>>(
      l1x, l2x, l1f, idx2, 1024, 256, 64, 64, 64, 128, XA);
  run_layer(L[3], XA, YB, XA, 65536, 0.f);
  run_layer(L[4], XA, YB, XA, 65536, 0.f);
  run_layer(L[5], XA, YB, XA, 65536, 0.f);
  maxpool_kernel<<<cdiv_h((long)1024 * 128, 256), 256, 0, stream>>>(XA, 64, 128, 1024, l2f);

  // ---- SA3: 256 -> 64, r=0.4, S=128, MLP[128,128,256] ----
  fps_kernel<<<BB, 1024, 0, stream>>>(l2x, 256, 64, fidx3);
  gather_xyz_kernel<<<cdiv_h(BB * 64, 256), 256, 0, stream>>>(l2x, fidx3, 256, 64, l3x);
  ball_query_kernel<<<cdiv_h(BB * 64, 64), 64, 0, stream>>>(l2x, l3x, 256, 64,
                                                            0.16f, 128, idx3);
  group_gather_kernel<<<cdiv_h((long)BB * 64 * 128, 256), 256, 0, stream>>>(
      l2x, l3x, l2f, idx3, 256, 64, 128, 128, 128, 192, XA);
  run_layer(L[6], XA, YB, XA, 32768, 0.f);
  run_layer(L[7], XA, YB, XA, 32768, 0.f);
  run_layer(L[8], XA, YB, XA, 32768, 0.f);
  maxpool_kernel<<<cdiv_h((long)256 * 256, 256), 256, 0, stream>>>(XA, 128, 256, 256, l3f);

  // ---- SA4 (group-all): MLP[256,256,512], maxpool over the 64 points ----
  sa_all_concat_kernel<<<cdiv_h(256, 256), 256, 0, stream>>>(l3x, l3f, 256, 256, 256, 320, XA);
  run_layer(L[9],  XA, YB, XA, 256, 0.f);
  run_layer(L[10], XA, YB, XA, 256, 0.f);
  run_layer(L[11], XA, YB, XA, 256, 0.f);
  maxpool_kernel<<<cdiv_h((long)4 * 512, 256), 256, 0, stream>>>(XA, 64, 512, 4, l4f);

  // ---- FP4: broadcast l4f + l3f -> MLP[256,256] ----
  fp4_concat_kernel<<<cdiv_h(256L * 768, 256), 256, 0, stream>>>(l4f, l3f, XA);
  run_layer(L[12], XA, YB, XA, 256, 0.f);
  run_layer(L[13], XA, YB, l3f2, 256, 0.f);

  // ---- FP3: interp l3 -> l2, MLP[256,256] ----
  fp_interp_kernel<<<cdiv_h(BB * 256, 128), 128, 0, stream>>>(
      l2x, l3x, l3f2, l2f, 256, 64, 256, 256, 128, 128, 384, XA);
  run_layer(L[14], XA, YB, XA, 1024, 0.f);
  run_layer(L[15], XA, YB, l2f2, 1024, 0.f);

  // ---- FP2: interp l2 -> l1, MLP[256,128] ----
  fp_interp_kernel<<<cdiv_h(BB * 1024, 128), 128, 0, stream>>>(
      l1x, l2x, l2f2, l1f, 1024, 256, 256, 256, 64, 64, 320, XA);
  run_layer(L[16], XA, YB, XA, 4096, 0.f);
  run_layer(L[17], XA, YB, l1f2, 4096, 0.f);

  // ---- FP1: interp l1 -> full res, MLP[128,128] ----
  fp_interp_kernel<<<cdiv_h(BB * NN, 128), 128, 0, stream>>>(
      xyz, l1x, l1f2, uf1, NN, 1024, 128, 128, 6, 6, 192, XA);
  run_layer(L[18], XA, YB, XA, 65536, 0.f);
  run_layer(L[19], XA, YB, XA, 65536, 0.f);

  // ---- conv1 / conv2 (leaky 0.2) ----
  run_layer(L[20], XA, YB, XA, 65536, 0.2f);
  run_layer(L[21], XA, YB, XA, 65536, 0.2f);

  // ---- conv3: plain GEMM (no BN), f32 out, scatter into 3-way split ----
  {
    int Kp = 128, Op = 192;
    wprep_kernel<<<cdiv_h((long)Op * Kp, 256), 256, 0, stream>>>(conv3W, 135, 128, Op, Kp, Wb);
    dim3 g(65536 / 64, Op / 64);
    gemm_bf16_kernel<<<g, 128, 0, stream>>>(XA, Wb, YB, Yf32, 65536, Kp, Op, nullptr);
    out_scatter_kernel<<<cdiv_h((long)BB * NN * 135, 256), 256, 0, stream>>>(Yf32, (float*)d_out);
  }
}
